// DepthSelfAttention_56590489092455
// MI455X (gfx1250) — matrix operand order repaired and checked by reading, not compiled, not run on hardware
//
#include <hip/hip_runtime.h>
#include <hip/hip_bf16.h>

// ---------------------------------------------------------------------------
// DepthSelfAttention for MI455X (gfx1250):
//   prepass: Wq/Wk/Wv/Wproj f32 -> f16 (Wk,Wv fused into one 1024-row Wkv)
//   Q  = x  @ Wq^T        (8192 x 2048 x 2048)   f16 WMMA, f32 accum
//   KV = kv @ Wkv^T       (65536 x 2048 x 1024)  kv gathered; K+V fused => the
//                          512MB kv_source stream is read ONCE, not twice
//   epilogue: rmsnorm + depth-RoPE + GQA softmax(8) -> y (f16)
//   out = y @ Wproj^T     (8192 x 2048 x 2048)
// GEMMs are software-pipelined with double-buffered LDS: the Tensor Data
// Mover stages tile i+1 (tensor_load_to_lds, s_wait_tensorcnt <= S keeps it
// in flight) while v_wmma_f32_16x16x32_f16 consumes tile i. f32 A operands
// are prefetched into registers one tile ahead and converted after compute.
// ---------------------------------------------------------------------------

typedef __attribute__((ext_vector_type(16))) _Float16 v16h;
typedef __attribute__((ext_vector_type(8)))  _Float16 v8h;
typedef __attribute__((ext_vector_type(4)))  _Float16 v4h;
typedef __attribute__((ext_vector_type(8)))  float    v8f;
typedef __attribute__((ext_vector_type(4)))  float    v4f;
typedef __attribute__((ext_vector_type(4)))  unsigned v4u;
typedef __attribute__((ext_vector_type(8)))  int      v8i_t;
typedef __attribute__((ext_vector_type(4)))  int      v4i_t;

#define LDS_STRIDE 48   // halves; 96B row stride; TDM pad: 16 DW data + 8 DW pad

// A-source modes
#define AMODE_F32 0   // A = f32 matrix (x)
#define AMODE_KV  1   // A row r: n=r>>3, d=r&7 ; d<7 -> depth_history, d==7 -> x
#define AMODE_F16 2   // A = f16 matrix (attention output y) -> staged via TDM

// ---------------------------------------------------------------------------
// TDM: async-load a [tile_rows x 32] f16 tile (row stride = kdim elements)
// from global into LDS at byte offset ldsOff, padding 8 DWORDs after every
// 16 DWORDs so LDS rows land on the 48-half stride.
// ---------------------------------------------------------------------------
__device__ __forceinline__ void tdm_load_tile_f16(
    const _Float16* gsrc, unsigned ldsOff,
    int kdim, int rows_remaining, int k_remaining, int tile_rows)
{
    unsigned long long ga = (unsigned long long)(const void*)gsrc;
    v4u g0;
    g0[0] = 1u;                                       // count=1, user mode
    g0[1] = ldsOff;                                   // lds_addr (bytes)
    g0[2] = (unsigned)(ga & 0xffffffffull);           // global_addr[31:0]
    g0[3] = (unsigned)((ga >> 32) & 0x01ffffffull)    // global_addr[56:32]
          | 0x80000000u;                              // type=2 ("image")
    unsigned dim0 = (unsigned)k_remaining;            // tensor_dim0 (elements)
    unsigned dim1 = (unsigned)rows_remaining;         // tensor_dim1 (rows)
    v8i_t g1;
    g1[0] = (int)((1u << 16)        // data_size = 1 -> 2 bytes
                | (1u << 20)        // pad_enable
                | (3u << 22)        // pad_interval: 16 DWORDs
                | (7u << 25));      // pad_amount:   8 DWORDs
    g1[1] = (int)((dim0 & 0xffffu) << 16);                       // dim0 lo16
    g1[2] = (int)((dim0 >> 16) | ((dim1 & 0xffffu) << 16));      // dim0 hi/dim1 lo
    g1[3] = (int)((dim1 >> 16) | (32u << 16));                   // dim1 hi, tile_dim0=32
    g1[4] = (int)(unsigned)tile_rows;                            // tile_dim1, tile_dim2=0
    g1[5] = (int)(unsigned)kdim;                                 // dim0_stride[31:0]
    g1[6] = 0;                                                   // stride hi, dim1_stride lo
    g1[7] = 0;
    v4i_t gz = {};
#if __clang_major__ >= 23
    v8i_t gz8 = {};
    __builtin_amdgcn_tensor_load_to_lds(g0, g1, gz, gz, gz8, 0);
#else
    __builtin_amdgcn_tensor_load_to_lds(g0, g1, gz, gz, 0);
#endif
}

template<int AMODE>
__global__ __launch_bounds__(256) void gemm_wmma_kernel(
    const float*    __restrict__ Ax,    // x (AMODE_F32 / AMODE_KV)
    const float*    __restrict__ Adh,   // depth_history (AMODE_KV)
    const _Float16* __restrict__ Ah,    // f16 A (AMODE_F16)
    const _Float16* __restrict__ W16,   // [Ncols][Kdim] f16 row-major
    float*          __restrict__ C,     // [Mrows][Ncols] f32
    int Mrows, int Ncols, int Kdim)
{
    __shared__ _Float16 As[2][128 * LDS_STRIDE];
    __shared__ _Float16 Bs[2][128 * LDS_STRIDE];

    const int tid     = threadIdx.x;
    const int lane    = tid & 31;
    const int wid     = tid >> 5;           // 8 waves
    const int rowBase = blockIdx.y * 128;
    const int colBase = blockIdx.x * 128;
    const int wm      = (wid >> 1) * 32;
    const int wn      = (wid & 1) * 64;
    const int lrow    = lane & 15;
    const int lhi     = lane >> 4;

    // LDS byte offsets (low 32 bits of generic LDS address = LDS offset)
    const unsigned asOff = (unsigned)(unsigned long long)(void*)&As[0][0];
    const unsigned bsOff = (unsigned)(unsigned long long)(void*)&Bs[0][0];
    const unsigned bufBytes = 128 * LDS_STRIDE * 2;

    // ---- pipeline helpers --------------------------------------------------
    // Issue TDM stage for K-step k0 into buffer `buf` (B tile; A too if f16).
    auto stage_tdm = [&](int buf, int k0) {
        int rb = colBase + wid * 16;
        tdm_load_tile_f16(W16 + (size_t)rb * Kdim + k0,
                          bsOff + (unsigned)buf * bufBytes
                                + (unsigned)(wid * 16 * LDS_STRIDE * 2),
                          Kdim, Ncols - rb, Kdim - k0, 16);
        if (AMODE == AMODE_F16) {
            int ra = rowBase + wid * 16;
            tdm_load_tile_f16(Ah + (size_t)ra * Kdim + k0,
                              asOff + (unsigned)buf * bufBytes
                                    + (unsigned)(wid * 16 * LDS_STRIDE * 2),
                              Kdim, Mrows - ra, Kdim - k0, 16);
        }
    };
    // Issue the 4 batched global f32 loads of the A tile at K-step k0.
    auto load_a = [&](int k0, v4f* val) {
        #pragma unroll
        for (int i = 0; i < 4; i++) {
            int c   = tid + 256 * i;       // 1024 chunks of 4 floats
            int row = c >> 3;
            int ko  = (c & 7) * 4;
            const float* p;
            if (AMODE == AMODE_KV) {
                int gr = rowBase + row;
                int nn = gr >> 3;
                int dd = gr & 7;
                p = (dd < 7) ? (Adh + ((size_t)nn * 7 + dd) * Kdim)
                             : (Ax  + (size_t)nn * Kdim);
            } else {
                p = Ax + (size_t)(rowBase + row) * Kdim;
            }
            val[i] = *(const v4f*)(p + k0 + ko);
        }
    };
    // Convert prefetched A registers to f16 and store into buffer `buf`.
    auto store_a = [&](int buf, const v4f* val) {
        _Float16* dst = &As[buf][0];
        #pragma unroll
        for (int i = 0; i < 4; i++) {
            int c   = tid + 256 * i;
            int row = c >> 3;
            int ko  = (c & 7) * 4;
            v4h hv;
            hv[0] = (_Float16)val[i][0]; hv[1] = (_Float16)val[i][1];
            hv[2] = (_Float16)val[i][2]; hv[3] = (_Float16)val[i][3];
            *(v4h*)(dst + row * LDS_STRIDE + ko) = hv;
        }
    };

    v8f acc[2][4] = {};
    v4f aval[4];

    // ---- prologue: stage tile 0 into buffer 0 ------------------------------
    const int nIter = Kdim >> 5;
    if (AMODE != AMODE_F16) load_a(0, aval);
    stage_tdm(0, 0);
    if (AMODE != AMODE_F16) store_a(0, aval);

    for (int i = 0; i < nIter; i++) {
        const int  buf     = i & 1;
        const int  k0      = i << 5;
        const bool hasNext = (i + 1 < nIter);

        // issue next tile's staging (into buf^1; safe: B2 of iter i-1 passed)
        if (hasNext) {
            if (AMODE != AMODE_F16) load_a(k0 + 32, aval);
            stage_tdm(buf ^ 1, k0 + 32);
        }
        // wait only for THIS tile's TDM ops; next tile's stay in flight
        if (hasNext) {
            if (AMODE == AMODE_F16) __builtin_amdgcn_s_wait_tensorcnt(2);
            else                    __builtin_amdgcn_s_wait_tensorcnt(1);
        } else {
            __builtin_amdgcn_s_wait_tensorcnt(0);
        }
        __syncthreads();            // B1: tile i staging visible to all waves

        // ---- fragments + WMMA on buffer `buf` -----------------------------
        const _Float16* asb = &As[buf][0];
        const _Float16* bsb = &Bs[buf][0];
        v16h a[2];
        #pragma unroll
        for (int mi = 0; mi < 2; mi++) {
            const _Float16* p = asb + (wm + mi * 16 + lrow) * LDS_STRIDE + lhi * 8;
            v8h lo = *(const v8h*)p;
            v8h hi = *(const v8h*)(p + 16);
            a[mi] = __builtin_shufflevector(lo, hi, 0,1,2,3,4,5,6,7,8,9,10,11,12,13,14,15);
        }
        v16h b[4];
        #pragma unroll
        for (int ni = 0; ni < 4; ni++) {
            const _Float16* p = bsb + (wn + ni * 16 + lrow) * LDS_STRIDE + lhi * 8;
            v8h lo = *(const v8h*)p;
            v8h hi = *(const v8h*)(p + 16);
            b[ni] = __builtin_shufflevector(lo, hi, 0,1,2,3,4,5,6,7,8,9,10,11,12,13,14,15);
        }
        #pragma unroll
        for (int mi = 0; mi < 2; mi++)
            #pragma unroll
            for (int ni = 0; ni < 4; ni++)
                acc[mi][ni] = __builtin_amdgcn_wmma_f32_16x16x32_f16(
                    false, a[mi], false, b[ni], (short)0, acc[mi][ni], false, false);

        // convert + store next A tile (loads had compute-phase latency cover)
        if (hasNext && AMODE != AMODE_F16) store_a(buf ^ 1, aval);

        __syncthreads();            // B2: all reads of `buf` done -> reusable
    }

    // ---- writeback: VGPR r -> M = r + 8*(lane>=16), N = lane%16 ------------
    #pragma unroll
    for (int mi = 0; mi < 2; mi++) {
        #pragma unroll
        for (int ni = 0; ni < 4; ni++) {
            int ncol = colBase + wn + ni * 16 + lrow;
            #pragma unroll
            for (int r = 0; r < 8; r++) {
                int mrow = rowBase + wm + mi * 16 + r + 8 * lhi;
                C[(size_t)mrow * Ncols + ncol] = acc[mi][ni][r];
            }
        }
    }
}

// ---------------------------------------------------------------------------
// f32 -> f16 weight conversion
// ---------------------------------------------------------------------------
__global__ __launch_bounds__(256) void f32_to_f16_kernel(
    const float* __restrict__ src, _Float16* __restrict__ dst, int n4)
{
    int idx = blockIdx.x * 256 + threadIdx.x;
    if (idx < n4) {
        v4f v = ((const v4f*)src)[idx];
        v4h h;
        h[0] = (_Float16)v[0]; h[1] = (_Float16)v[1];
        h[2] = (_Float16)v[2]; h[3] = (_Float16)v[3];
        ((v4h*)dst)[idx] = h;
    }
}

// ---------------------------------------------------------------------------
// Per-token attention epilogue. One block (8 waves) per token n.
// KVb row = n*8+d (stride 1024): cols [kvh*128 .. ) = K, +512 = V.
// ---------------------------------------------------------------------------
__global__ __launch_bounds__(256) void attn_kernel(
    const float* __restrict__ Q,    // [8192][2048]
    const float* __restrict__ KVb,  // [8192*8][1024]
    const float* __restrict__ qg,   // [16]
    _Float16*    __restrict__ Y)    // [8192][2048] f16
{
    __shared__ float kb[4][8][128];
    __shared__ float vb[4][8][128];

    const int n    = blockIdx.x;
    const int tid  = threadIdx.x;
    const int lane = tid & 31;
    const int w    = tid >> 5;

    const float LOGB = 9.210340371976184f;   // ln(10000)
    const float f0 = __expf(-((float)lane        * (1.0f / 64.0f)) * LOGB);
    const float f1 = __expf(-((float)(lane + 32) * (1.0f / 64.0f)) * LOGB);

    #pragma unroll
    for (int t = 0; t < 4; t++) {
        int kvidx = w * 4 + t;
        int kvh   = kvidx >> 3;
        int d     = kvidx & 7;
        const float* kr = KVb + ((size_t)n * 8 + d) * 1024 + kvh * 128;
        float k0 = kr[lane], k1 = kr[lane + 32], k2 = kr[lane + 64], k3 = kr[lane + 96];
        float ss = k0 * k0 + k1 * k1 + k2 * k2 + k3 * k3;
        #pragma unroll
        for (int o = 16; o > 0; o >>= 1) ss += __shfl_xor(ss, o, 32);
        float rinv = rsqrtf(ss * (1.0f / 128.0f) + 1.1920929e-07f);
        k0 *= rinv; k1 *= rinv; k2 *= rinv; k3 *= rinv;
        float pos = (float)d, rpos = (float)(15 - d);
        float s0, c0, s1, c1, rs0, rc0, rs1, rc1;
        __sincosf(pos * f0,  &s0,  &c0);
        __sincosf(pos * f1,  &s1,  &c1);
        __sincosf(rpos * f0, &rs0, &rc0);
        __sincosf(rpos * f1, &rs1, &rc1);
        kb[kvh][d][lane]      =  k0 * c0  + k2 * s0;
        kb[kvh][d][lane + 32] =  k1 * c1  + k3 * s1;
        kb[kvh][d][lane + 64] = -k0 * rs0 + k2 * rc0;
        kb[kvh][d][lane + 96] = -k1 * rs1 + k3 * rc1;
        const float* vr = kr + 512;
        vb[kvh][d][lane]      = vr[lane];
        vb[kvh][d][lane + 32] = vr[lane + 32];
        vb[kvh][d][lane + 64] = vr[lane + 64];
        vb[kvh][d][lane + 96] = vr[lane + 96];
    }
    __syncthreads();

    #pragma unroll
    for (int t = 0; t < 2; t++) {
        int h   = w * 2 + t;
        int kvh = h >> 2;
        const float* qr = Q + (size_t)n * 2048 + h * 128;
        float q0 = qr[lane], q1 = qr[lane + 32], q2 = qr[lane + 64], q3 = qr[lane + 96];
        float ss = q0 * q0 + q1 * q1 + q2 * q2 + q3 * q3;
        #pragma unroll
        for (int o = 16; o > 0; o >>= 1) ss += __shfl_xor(ss, o, 32);
        float rinv = rsqrtf(ss * (1.0f / 128.0f) + 1.1920929e-07f);
        q0 *= rinv; q1 *= rinv; q2 *= rinv; q3 *= rinv;
        float s0, c0, s1, c1, rs0, rc0, rs1, rc1;
        __sincosf(7.0f * f0, &s0,  &c0);     // pos  = depth-1 = 7
        __sincosf(7.0f * f1, &s1,  &c1);
        __sincosf(8.0f * f0, &rs0, &rc0);    // rpos = 15-7    = 8
        __sincosf(8.0f * f1, &rs1, &rc1);
        float g  = qg[h];
        float n0 = ( q0 * c0  + q2 * s0)  * g;
        float n1 = ( q1 * c1  + q3 * s1)  * g;
        float n2 = (-q0 * rs0 + q2 * rc0) * g;
        float n3 = (-q1 * rs1 + q3 * rc1) * g;

        float sc[8];
        #pragma unroll
        for (int d = 0; d < 8; d++) {
            float dot = n0 * kb[kvh][d][lane]      + n1 * kb[kvh][d][lane + 32]
                      + n2 * kb[kvh][d][lane + 64] + n3 * kb[kvh][d][lane + 96];
            #pragma unroll
            for (int o = 16; o > 0; o >>= 1) dot += __shfl_xor(dot, o, 32);
            sc[d] = dot * 0.08838834764831843f;   // 1/sqrt(128)
        }
        float mx = sc[0];
        #pragma unroll
        for (int d = 1; d < 8; d++) mx = fmaxf(mx, sc[d]);
        float es = 0.0f;
        #pragma unroll
        for (int d = 0; d < 8; d++) { sc[d] = __expf(sc[d] - mx); es += sc[d]; }
        float inv = 1.0f / es;
        float y0 = 0.f, y1 = 0.f, y2 = 0.f, y3 = 0.f;
        #pragma unroll
        for (int d = 0; d < 8; d++) {
            float wgt = sc[d] * inv;
            y0 += wgt * vb[kvh][d][lane];
            y1 += wgt * vb[kvh][d][lane + 32];
            y2 += wgt * vb[kvh][d][lane + 64];
            y3 += wgt * vb[kvh][d][lane + 96];
        }
        _Float16* yr = Y + (size_t)n * 2048 + h * 128;
        yr[lane]      = (_Float16)y0;
        yr[lane + 32] = (_Float16)y1;
        yr[lane + 64] = (_Float16)y2;
        yr[lane + 96] = (_Float16)y3;
    }
}

extern "C" void kernel_launch(void* const* d_in, const int* in_sizes, int n_in,
                              void* d_out, int out_size, void* d_ws, size_t ws_size,
                              hipStream_t stream) {
    const float* x   = (const float*)d_in[0];   // [4,2048,2048]
    const float* dh  = (const float*)d_in[1];   // [4,2048,7,2048]
    const float* Wq  = (const float*)d_in[2];   // [2048,2048]
    const float* Wk  = (const float*)d_in[3];   // [512,2048]
    const float* Wv  = (const float*)d_in[4];   // [512,2048]
    const float* Wp  = (const float*)d_in[5];   // [2048,2048]
    const float* qg  = (const float*)d_in[6];   // [16]
    float* out = (float*)d_out;                 // [4,2048,2048]

    char* ws = (char*)d_ws;
    float*    Qb    = (float*)ws;                              //  64 MB
    float*    KVb   = (float*)(ws + ((size_t)64  << 20));      // 256 MB
    _Float16* Yb    = (_Float16*)(ws + ((size_t)320 << 20));   //  32 MB
    _Float16* Wq16  = (_Float16*)(ws + ((size_t)352 << 20));   //   8 MB
    _Float16* Wkv16 = (_Float16*)(ws + ((size_t)360 << 20));   //   4 MB
    _Float16* Wp16  = (_Float16*)(ws + ((size_t)364 << 20));   //   8 MB

    dim3 blk(256);

    // ---- weight prepass: f32 -> f16 (Wk,Wv concatenated into Wkv16) --------
    f32_to_f16_kernel<<<dim3(4096), blk, 0, stream>>>(Wq, Wq16, 1024 * 1024);
    f32_to_f16_kernel<<<dim3(1024), blk, 0, stream>>>(Wk, Wkv16, 256 * 1024);
    f32_to_f16_kernel<<<dim3(1024), blk, 0, stream>>>(Wv, Wkv16 + 512 * 2048, 256 * 1024);
    f32_to_f16_kernel<<<dim3(4096), blk, 0, stream>>>(Wp, Wp16, 1024 * 1024);

    // ---- Q = x @ Wq^T : M=8192, N=2048, K=2048 -----------------------------
    gemm_wmma_kernel<AMODE_F32><<<dim3(16, 64), blk, 0, stream>>>(
        x, nullptr, nullptr, Wq16, Qb, 8192, 2048, 2048);

    // ---- KV = kv @ Wkv^T : M=65536, N=1024, K=2048 (kv read once) ----------
    gemm_wmma_kernel<AMODE_KV><<<dim3(8, 512), blk, 0, stream>>>(
        x, dh, nullptr, Wkv16, KVb, 65536, 1024, 2048);

    // ---- attention epilogue -> y (f16) -------------------------------------
    attn_kernel<<<dim3(8192), blk, 0, stream>>>(Qb, KVb, qg, Yb);

    // ---- out = y @ Wproj^T : M=8192, N=2048, K=2048 ------------------------
    gemm_wmma_kernel<AMODE_F16><<<dim3(16, 64), blk, 0, stream>>>(
        nullptr, nullptr, Yb, Wp16, out, 8192, 2048, 2048);
}
